// QM9GraphRegMLA_12962211299662
// MI455X (gfx1250) — matrix operand order repaired and checked by reading, not compiled, not run on hardware
//
#include <hip/hip_runtime.h>
#include <math.h>

// ---------------------------------------------------------------------------
// QM9 graph regression w/ MLA attention for gfx1250 (MI455X, wave32, WMMA).
// Dense GEMMs: TDM (tensor_load_to_lds) stages A tile + BT weight into LDS
// (bank-conflict-free via descriptor padding), waves consume via ds_load_b128
// and v_wmma_f32_16x16x32_f16. Edge scatter uses hardware fp32 global atomics.
// ---------------------------------------------------------------------------

typedef __attribute__((ext_vector_type(8)))  _Float16 v8h;
typedef __attribute__((ext_vector_type(16))) _Float16 v16h;
typedef __attribute__((ext_vector_type(8)))  float    v8f;
typedef __attribute__((ext_vector_type(4)))  unsigned int v4u;
typedef __attribute__((ext_vector_type(4)))  int      v4i;
typedef __attribute__((ext_vector_type(8)))  int      v8i;

#define HDIM   128
#define HEADS  8
#define HD     16
#define OUTD   19
#define LPAD   8   // LDS row padding in halfs (TDM pad_amount = 4 DWORDs = 16B)

__device__ __forceinline__ void atomAddF(float* p, float v) {
  unsafeAtomicAdd(p, v);   // -> global_atomic_add_f32
}

__device__ __forceinline__ void atomMaxF(float* addr, float val) {
  int* ia  = (int*)addr;
  int  old = __float_as_int(*addr);
  while (val > __int_as_float(old)) {
    int assumed = old;
    old = atomicCAS(ia, assumed, __float_as_int(val));
    if (old == assumed) break;
  }
}

// --------- TDM: DMA a 2D tile [rows x K halfs] global->LDS, row-padded ------
// D# per CDNA5 ISA (8.3/8.4): count=1, type=2, data_size=2B,
// pad_enable, pad_interval = one row of DWORDs, pad_amount = 4 DWORDs.
// Rows beyond availRows are zero-filled by TDM OOB handling.
__device__ __forceinline__ void tdm_load_2d(_Float16* ldsDst, const _Float16* gsrc,
                                            int K, int availRows, int tileRows) {
  unsigned long long ga = (unsigned long long)(uintptr_t)gsrc;
  unsigned la = (unsigned)(uintptr_t)ldsDst;         // low 32b of flat = LDS offset
  int ic = (31 - __builtin_clz((unsigned)(K >> 1))) - 1;  // interval code: K/2 DWORDs
  union { unsigned w[4]; v4u v; } g0 = {};
  g0.w[0] = 1u;                                             // count=1 (valid)
  g0.w[1] = la;                                             // lds_addr
  g0.w[2] = (unsigned)ga;                                   // global_addr[31:0]
  g0.w[3] = (unsigned)((ga >> 32) & 0x1FFFFFFull) | (2u << 30); // addr[56:32]|type=2
  union { unsigned w[8]; v8i v; } g1 = {};
  g1.w[0] = (1u << 16)            // data_size = 2 bytes
          | (1u << 20)            // pad_enable
          | ((unsigned)ic << 22)  // pad_interval
          | (3u << 25);           // pad_amount = 4 DWORDs (16B, keeps b128 align)
  g1.w[1] = ((unsigned)K & 0xFFFFu) << 16;                            // tdim0 lo
  g1.w[2] = ((unsigned)K >> 16) | (((unsigned)availRows & 0xFFFFu) << 16); // tdim0 hi|tdim1 lo
  g1.w[3] = ((unsigned)availRows >> 16) | (((unsigned)K & 0xFFFFu) << 16); // tdim1 hi|tile_dim0
  g1.w[4] = (unsigned)tileRows & 0xFFFFu;                             // tile_dim1 (tile_dim2=0)
  g1.w[5] = (unsigned)K;                                              // tdim0_stride lo
  g1.w[6] = 0; g1.w[7] = 0;
  v4i z4 = {0,0,0,0};
#if defined(__clang_major__) && (__clang_major__ >= 23)
  v8i z8 = {0,0,0,0,0,0,0,0};
  __builtin_amdgcn_tensor_load_to_lds(g0.v, g1.v, z4, z4, z8, 0);
#else
  __builtin_amdgcn_tensor_load_to_lds(g0.v, g1.v, z4, z4, 0);
#endif
}

// ---------------- elementwise / init ----------------
__global__ void k_fill(float* p, float v, int n) {
  int i = blockIdx.x * blockDim.x + threadIdx.x;
  if (i < n) p[i] = v;
}

// transpose fp32 [K x Nout] weight -> fp16 [Nout x K] (column-major for WMMA B)
__global__ void k_transpose_f16(const float* __restrict__ w, _Float16* __restrict__ wT,
                                int K, int Nout) {
  int i = blockIdx.x * blockDim.x + threadIdx.x;
  if (i >= K * Nout) return;
  int n = i / K, k = i % K;
  wT[i] = (_Float16)w[(size_t)k * Nout + n];
}

// h = node_emb[x[:,0]]  (fp16 copy for WMMA A)
__global__ void k_embed(const int* __restrict__ x, const float* __restrict__ emb,
                        _Float16* __restrict__ hf16, int N) {
  int i = blockIdx.x * blockDim.x + threadIdx.x;
  if (i >= N * HDIM) return;
  int node = i >> 7, c = i & 127;
  int tok = x[node * 11];
  hf16[i] = (_Float16)emb[(size_t)tok * HDIM + c];
}

__global__ void k_deg_edge(const int* __restrict__ dst, float* __restrict__ deg, int E) {
  int e = blockIdx.x * blockDim.x + threadIdx.x;
  if (e < E) atomAddF(&deg[dst[e]], 1.0f);
}

__global__ void k_dinv(const float* __restrict__ deg, float* __restrict__ dinv, int N) {
  int i = blockIdx.x * blockDim.x + threadIdx.x;
  if (i < N) { float d = deg[i]; dinv[i] = d > 0.f ? rsqrtf(d) : 0.f; }
}

// ------- WMMA GEMM: out[M x Nout] = A[M x K](f16) @ B (BT[Nout x K] f16) ----
// TDM stages the block's A tile and whole BT into LDS (padded rows), then
// 8 waves/block each compute one 16x16 tile from LDS fragments.
__global__ void k_gemm_wmma(const _Float16* __restrict__ A, const _Float16* __restrict__ BT,
                            const float* __restrict__ bias,
                            float* __restrict__ o32, _Float16* __restrict__ o16,
                            int M, int K, int Nout, int NT, int relu) {
  extern __shared__ _Float16 lds[];
  const int RT = 8 / NT;
  const int SA = K + LPAD;                  // padded LDS row stride (halfs)
  _Float16* ldsA = lds;                     // RT*16 rows
  _Float16* ldsB = lds + RT * 16 * SA;      // Nout rows
  const int blockRow = blockIdx.x * RT * 16;

  if (threadIdx.x < 32) {                   // wave 0 issues both DMAs
    int avail = M - blockRow; if (avail < 0) avail = 0;
    tdm_load_2d(ldsA, A + (size_t)blockRow * K, K, avail, RT * 16);
    tdm_load_2d(ldsB, BT, K, Nout, Nout);
    __builtin_amdgcn_s_wait_tensorcnt(0);
  }
  __syncthreads();

  const int wid  = threadIdx.x >> 5;
  const int lane = threadIdx.x & 31;
  const int rowBase = blockRow + (wid / NT) * 16;
  if (rowBase >= M) return;                 // wave-uniform: EXEC stays all-ones
  const int colBase = (wid % NT) * 16;
  const int n  = lane & 15;
  const int hi = lane >> 4;                 // lane half-group

  const _Float16* Arow = ldsA + (size_t)((wid / NT) * 16 + n) * SA;
  const _Float16* Brow = ldsB + (size_t)(colBase + n) * SA;

  v8f acc = {0.f,0.f,0.f,0.f,0.f,0.f,0.f,0.f};
  const int kTiles = K >> 5;
  for (int kt = 0; kt < kTiles; ++kt) {
    // A 16x32 f16: lane half 0 -> K {0..7,16..23}; half 1 -> K {8..15,24..31}
    const int kbA = kt * 32 + hi * 8;
    v8h a0 = *(const v8h*)(Arow + kbA);
    v8h a1 = *(const v8h*)(Arow + kbA + 16);
    v16h av = __builtin_shufflevector(a0, a1, 0,1,2,3,4,5,6,7,8,9,10,11,12,13,14,15);
    // B 32x16 f16: lane half 0 -> K {0..7,16..23}; half 1 -> K {8..15,24..31}
    const int kbB = kt * 32 + hi * 8;
    v8h b0 = *(const v8h*)(Brow + kbB);
    v8h b1 = *(const v8h*)(Brow + kbB + 16);
    v16h bv = __builtin_shufflevector(b0, b1, 0,1,2,3,4,5,6,7,8,9,10,11,12,13,14,15);
    acc = __builtin_amdgcn_wmma_f32_16x16x32_f16(false, av, false, bv,
                                                 (short)0, acc, false, false);
  }

  const float bval = bias ? bias[colBase + n] : 0.0f;
  #pragma unroll
  for (int r = 0; r < 8; ++r) {
    int m = rowBase + r + hi * 8;
    if (m < M) {
      float v = acc[r] + bval;
      if (relu) v = fmaxf(v, 0.0f);
      size_t off = (size_t)m * Nout + colBase + n;
      if (o32) o32[off] = v;
      if (o16) o16[off] = (_Float16)v;
    }
  }
}

// ---------------- GCN aggregation ----------------
__global__ void k_gcn_self(const float* __restrict__ hl, const float* __restrict__ dinv,
                           float* __restrict__ h1, int N) {
  int i = blockIdx.x * blockDim.x + threadIdx.x;
  if (i >= N * HDIM) return;
  int node = i >> 7;
  float d = dinv[node];
  h1[i] = hl[i] * d * d;
}

__global__ void k_gcn_edge(const int* __restrict__ src, const int* __restrict__ dst,
                           const float* __restrict__ hl, const float* __restrict__ dinv,
                           float* __restrict__ h1, int E) {
  int w = (blockIdx.x * blockDim.x + threadIdx.x) >> 5;
  int lane = threadIdx.x & 31;
  if (w >= E) return;
  int s = src[w], d = dst[w];
  float nrm = dinv[s] * dinv[d];
  float4 hv = ((const float4*)(hl + (size_t)s * HDIM))[lane];
  float* op = h1 + (size_t)d * HDIM + lane * 4;
  atomAddF(op + 0, nrm * hv.x);
  atomAddF(op + 1, nrm * hv.y);
  atomAddF(op + 2, nrm * hv.z);
  atomAddF(op + 3, nrm * hv.w);
}

__global__ void k_gcn_finish(float* __restrict__ h1, const float* __restrict__ b,
                             _Float16* __restrict__ h1h, int N) {
  int i = blockIdx.x * blockDim.x + threadIdx.x;
  if (i >= N * HDIM) return;
  float v = fmaxf(h1[i] + b[i & 127], 0.0f);
  h1[i]  = v;
  h1h[i] = (_Float16)v;
}

// ---------------- MLA edge softmax ----------------
__global__ void k_score(const float* __restrict__ q, const float* __restrict__ k,
                        const int* __restrict__ src, const int* __restrict__ dst,
                        float* __restrict__ score, float* __restrict__ m, int E) {
  int idx = blockIdx.x * blockDim.x + threadIdx.x;
  if (idx >= E * HEADS) return;
  int e = idx >> 3, h = idx & 7;
  int s = src[e], d = dst[e];
  const float4* qp = (const float4*)(q + (size_t)d * HDIM + h * HD);
  const float4* kp = (const float4*)(k + (size_t)s * HDIM + h * HD);
  float acc = 0.f;
  #pragma unroll
  for (int j = 0; j < 4; ++j) {
    float4 a = qp[j], b = kp[j];
    acc += a.x*b.x + a.y*b.y + a.z*b.z + a.w*b.w;
  }
  acc *= 0.25f;                        // 1/sqrt(HD=16)
  score[idx] = acc;
  atomMaxF(&m[d * HEADS + h], acc);
}

__global__ void k_mfix(float* __restrict__ m, int n) {
  int i = blockIdx.x * blockDim.x + threadIdx.x;
  if (i < n && m[i] == -INFINITY) m[i] = 0.0f;
}

__global__ void k_ex(const int* __restrict__ dst, const float* __restrict__ m,
                     float* __restrict__ score, float* __restrict__ den, int E) {
  int idx = blockIdx.x * blockDim.x + threadIdx.x;
  if (idx >= E * HEADS) return;
  int e = idx >> 3, h = idx & 7;
  int d = dst[e];
  float ex = __expf(score[idx] - m[d * HEADS + h]);
  score[idx] = ex;
  atomAddF(&den[d * HEADS + h], ex);
}

__global__ void k_agg(const int* __restrict__ src, const int* __restrict__ dst,
                      const float* __restrict__ ex, const float* __restrict__ den,
                      const float* __restrict__ v, float* __restrict__ agg, int E) {
  int idx = blockIdx.x * blockDim.x + threadIdx.x;
  if (idx >= E * HEADS) return;
  int e = idx >> 3, h = idx & 7;
  int s = src[e], d = dst[e];
  float alpha = ex[idx] / fmaxf(den[d * HEADS + h], 1e-16f);
  const float4* vp = (const float4*)(v + (size_t)s * HDIM + h * HD);
  float* ap = agg + (size_t)d * HDIM + h * HD;
  #pragma unroll
  for (int j = 0; j < 4; ++j) {
    float4 vv = vp[j];
    atomAddF(ap + j*4 + 0, alpha * vv.x);
    atomAddF(ap + j*4 + 1, alpha * vv.y);
    atomAddF(ap + j*4 + 2, alpha * vv.z);
    atomAddF(ap + j*4 + 3, alpha * vv.w);
  }
}

__global__ void k_f32_to_f16(const float* __restrict__ in, _Float16* __restrict__ out, int n) {
  int i = blockIdx.x * blockDim.x + threadIdx.x;
  if (i < n) out[i] = (_Float16)in[i];
}

// ------- LayerNorm + residual + relu + pooling (one wave per node) ----------
__global__ void k_ln_pool(const float* __restrict__ attn, const float* __restrict__ res,
                          const float* __restrict__ g, const float* __restrict__ b,
                          const int* __restrict__ batch, float* __restrict__ pooled, int N) {
  int w = (blockIdx.x * blockDim.x + threadIdx.x) >> 5;
  int lane = threadIdx.x & 31;
  if (w >= N) return;
  const float* a = attn + (size_t)w * HDIM;
  const float* r = res  + (size_t)w * HDIM;
  int c = lane * 4;
  float4 x;
  x.x = a[c+0] + r[c+0];  x.y = a[c+1] + r[c+1];
  x.z = a[c+2] + r[c+2];  x.w = a[c+3] + r[c+3];
  float s = x.x + x.y + x.z + x.w;
  #pragma unroll
  for (int off = 16; off > 0; off >>= 1) s += __shfl_xor(s, off, 32);
  float mu = s * (1.0f / HDIM);
  float4 dv; dv.x = x.x - mu; dv.y = x.y - mu; dv.z = x.z - mu; dv.w = x.w - mu;
  float vs = dv.x*dv.x + dv.y*dv.y + dv.z*dv.z + dv.w*dv.w;
  #pragma unroll
  for (int off = 16; off > 0; off >>= 1) vs += __shfl_xor(vs, off, 32);
  float rstd = rsqrtf(vs * (1.0f / HDIM) + 1e-5f);
  float* pp = pooled + (size_t)batch[w] * HDIM + c;
  atomAddF(pp + 0, fmaxf(dv.x * rstd * g[c+0] + b[c+0], 0.0f));
  atomAddF(pp + 1, fmaxf(dv.y * rstd * g[c+1] + b[c+1], 0.0f));
  atomAddF(pp + 2, fmaxf(dv.z * rstd * g[c+2] + b[c+2], 0.0f));
  atomAddF(pp + 3, fmaxf(dv.w * rstd * g[c+3] + b[c+3], 0.0f));
}

// ---------------- tiny FC: out[G x 19] = pooled @ fc_w + fc_b ----------------
__global__ void k_fc(const float* __restrict__ pooled, const float* __restrict__ fw,
                     const float* __restrict__ fb, float* __restrict__ out, int G) {
  int idx = blockIdx.x * blockDim.x + threadIdx.x;
  if (idx >= G * OUTD) return;
  int g = idx / OUTD, o = idx % OUTD;
  const float* p = pooled + (size_t)g * HDIM;
  float acc = fb[o];
  for (int k = 0; k < HDIM; ++k) acc += p[k] * fw[k * OUTD + o];
  out[idx] = acc;
}

// ---------------------------------------------------------------------------
static inline size_t alignUp(size_t x) { return (x + 255) & ~(size_t)255; }

extern "C" void kernel_launch(void* const* d_in, const int* in_sizes, int n_in,
                              void* d_out, int out_size, void* d_ws, size_t ws_size,
                              hipStream_t stream) {
  const int* x    = (const int*)d_in[0];
  const int* ei   = (const int*)d_in[1];
  const int* batc = (const int*)d_in[2];
  const float* node_emb = (const float*)d_in[3];
  const float* gcn_w = (const float*)d_in[4];  const float* gcn_b = (const float*)d_in[5];
  const float* qd_w  = (const float*)d_in[6];  const float* qd_b  = (const float*)d_in[7];
  const float* qu_w  = (const float*)d_in[8];  const float* qu_b  = (const float*)d_in[9];
  const float* kvd_w = (const float*)d_in[10]; const float* kvd_b = (const float*)d_in[11];
  const float* ku_w  = (const float*)d_in[12]; const float* ku_b  = (const float*)d_in[13];
  const float* vu_w  = (const float*)d_in[14]; const float* vu_b  = (const float*)d_in[15];
  const float* ow    = (const float*)d_in[16]; const float* ob    = (const float*)d_in[17];
  const float* ln_g  = (const float*)d_in[18]; const float* ln_b  = (const float*)d_in[19];
  const float* fc_w  = (const float*)d_in[20]; const float* fc_b  = (const float*)d_in[21];

  const int N = in_sizes[2];          // 50000
  const int E = in_sizes[1] / 2;      // 800000
  const int G = out_size / OUTD;      // 2048
  const int* src = ei;
  const int* dst = ei + E;

  // ---- workspace layout ----
  char* ws = (char*)d_ws;  size_t off = 0;
  auto take = [&](size_t bytes) { char* p = ws + off; off += alignUp(bytes); return p; };
  _Float16* gcn_wT = (_Float16*)take(128*128*2);
  _Float16* qd_wT  = (_Float16*)take(32*128*2);
  _Float16* qu_wT  = (_Float16*)take(128*32*2);
  _Float16* kvd_wT = (_Float16*)take(32*128*2);
  _Float16* ku_wT  = (_Float16*)take(128*32*2);
  _Float16* vu_wT  = (_Float16*)take(128*32*2);
  _Float16* owT    = (_Float16*)take(128*128*2);
  _Float16* hf16   = (_Float16*)take((size_t)N*HDIM*2);
  _Float16* h1h    = (_Float16*)take((size_t)N*HDIM*2);
  _Float16* cqh    = (_Float16*)take((size_t)N*32*2);
  _Float16* ckvh   = (_Float16*)take((size_t)N*32*2);
  _Float16* aggh   = (_Float16*)take((size_t)N*HDIM*2);
  float* h1    = (float*)take((size_t)N*HDIM*4);
  float* hlagg = (float*)take((size_t)N*HDIM*4);   // hl, later reused as agg
  float* qattn = (float*)take((size_t)N*HDIM*4);   // q, later reused as attn out
  float* kbuf  = (float*)take((size_t)N*HDIM*4);
  float* vbuf  = (float*)take((size_t)N*HDIM*4);
  float* score = (float*)take((size_t)E*HEADS*4);  // then ex, in place
  float* mbuf  = (float*)take((size_t)N*HEADS*4);
  float* den   = (float*)take((size_t)N*HEADS*4);
  float* deg   = (float*)take((size_t)N*4);
  float* dinv  = (float*)take((size_t)N*4);
  float* pooled= (float*)take((size_t)G*HDIM*4);
  (void)ws_size; (void)n_in;

  const int TB = 256;
  auto blks = [](long n, int t) { return (int)((n + t - 1) / t); };
  auto gemm = [&](const _Float16* A, const _Float16* BT, const float* bias,
                  float* o32, _Float16* o16, int M, int K, int Nout, int relu) {
    int NT = Nout / 16, RT = 8 / NT, rows = RT * 16;
    size_t shb = (size_t)((RT * 16 + Nout) * (K + LPAD)) * 2;   // padded LDS tiles
    k_gemm_wmma<<<blks(M, rows), 256, shb, stream>>>(A, BT, bias, o32, o16, M, K, Nout, NT, relu);
  };

  // 1) weight prep (f16, transposed to [Nout x K])
  k_transpose_f16<<<blks(128*128, TB), TB, 0, stream>>>(gcn_w, gcn_wT, 128, 128);
  k_transpose_f16<<<blks(128*32,  TB), TB, 0, stream>>>(qd_w,  qd_wT,  128, 32);
  k_transpose_f16<<<blks(32*128,  TB), TB, 0, stream>>>(qu_w,  qu_wT,  32, 128);
  k_transpose_f16<<<blks(128*32,  TB), TB, 0, stream>>>(kvd_w, kvd_wT, 128, 32);
  k_transpose_f16<<<blks(32*128,  TB), TB, 0, stream>>>(ku_w,  ku_wT,  32, 128);
  k_transpose_f16<<<blks(32*128,  TB), TB, 0, stream>>>(vu_w,  vu_wT,  32, 128);
  k_transpose_f16<<<blks(128*128, TB), TB, 0, stream>>>(ow,    owT,    128, 128);

  // 2) embedding + degree
  k_embed<<<blks((long)N*HDIM, TB), TB, 0, stream>>>(x, node_emb, hf16, N);
  k_fill<<<blks(N, TB), TB, 0, stream>>>(deg, 1.0f, N);            // self loop
  k_deg_edge<<<blks(E, TB), TB, 0, stream>>>(dst, deg, E);
  k_dinv<<<blks(N, TB), TB, 0, stream>>>(deg, dinv, N);

  // 3) GCN: hl = h @ gcn_w (WMMA), then normalized scatter-sum, bias, relu
  gemm(hf16, gcn_wT, nullptr, hlagg, nullptr, N, 128, 128, 0);
  k_gcn_self<<<blks((long)N*HDIM, TB), TB, 0, stream>>>(hlagg, dinv, h1, N);
  k_gcn_edge<<<blks((long)E*32, TB), TB, 0, stream>>>(src, dst, hlagg, dinv, h1, E);
  k_gcn_finish<<<blks((long)N*HDIM, TB), TB, 0, stream>>>(h1, gcn_b, h1h, N);

  // 4) MLA projections (all WMMA)
  gemm(h1h,  qd_wT,  qd_b,  nullptr, cqh,  N, 128, 32, 0);
  gemm(cqh,  qu_wT,  qu_b,  qattn,   nullptr, N, 32, 128, 0);      // q
  gemm(h1h,  kvd_wT, kvd_b, nullptr, ckvh, N, 128, 32, 0);
  gemm(ckvh, ku_wT,  ku_b,  kbuf,    nullptr, N, 32, 128, 0);      // k
  gemm(ckvh, vu_wT,  vu_b,  vbuf,    nullptr, N, 32, 128, 0);      // v

  // 5) edge softmax attention
  k_fill<<<blks((long)N*HEADS, TB), TB, 0, stream>>>(mbuf, -INFINITY, N*HEADS);
  k_fill<<<blks((long)N*HEADS, TB), TB, 0, stream>>>(den, 0.0f, N*HEADS);
  k_score<<<blks((long)E*HEADS, TB), TB, 0, stream>>>(qattn, kbuf, src, dst, score, mbuf, E);
  k_mfix<<<blks((long)N*HEADS, TB), TB, 0, stream>>>(mbuf, N*HEADS);
  k_ex<<<blks((long)E*HEADS, TB), TB, 0, stream>>>(dst, mbuf, score, den, E);
  k_fill<<<blks((long)N*HDIM, TB), TB, 0, stream>>>(hlagg, 0.0f, N*HDIM);   // agg := 0
  k_agg<<<blks((long)E*HEADS, TB), TB, 0, stream>>>(src, dst, score, den, vbuf, hlagg, E);

  // 6) output projection (WMMA), LN + residual + relu + pooled segment-sum
  k_f32_to_f16<<<blks((long)N*HDIM, TB), TB, 0, stream>>>(hlagg, aggh, N*HDIM);
  gemm(aggh, owT, ob, qattn, nullptr, N, 128, 128, 0);             // attn out (reuse q buf)
  k_fill<<<blks((long)G*HDIM, TB), TB, 0, stream>>>(pooled, 0.0f, G*HDIM);
  k_ln_pool<<<blks((long)N*32, TB), TB, 0, stream>>>(qattn, h1, ln_g, ln_b, batc, pooled, N);

  // 7) final FC -> d_out
  k_fc<<<blks((long)G*OUTD, TB), TB, 0, stream>>>(pooled, fc_w, fc_b, (float*)d_out, G);
}
